// Block_data_51874615001686
// MI455X (gfx1250) — matrix operand order repaired and checked by reading, not compiled, not run on hardware
//
#include <hip/hip_runtime.h>
#include <hip/hip_bf16.h>

// ---------------- model constants ----------------
#define B_   2
#define N_   1024
#define DIM_ 768
#define H_   12
#define HD_  64
#define MLP_ 3072
// SCALE = HD^-0.5 = 0.125 exactly

typedef __attribute__((ext_vector_type(16))) __bf16 v16bf;
typedef __attribute__((ext_vector_type(8)))  float  v8f;
typedef __attribute__((ext_vector_type(4)))  float  f4;

union BF16x16 { f4 f[2]; v16bf v; };
static_assert(sizeof(v16bf) == 32, "v16bf must be 32 bytes");

// ---- CDNA5 async global->LDS copy (ASYNCcnt path), with sync fallback ----
#if __has_builtin(__builtin_amdgcn_global_load_async_to_lds_b128) && \
    __has_builtin(__builtin_amdgcn_s_wait_asynccnt)
#define USE_ASYNC 1
#else
#define USE_ASYNC 0
#endif

__device__ __forceinline__ void async_copy16(const __bf16* __restrict__ g, __bf16* l)
{
#if USE_ASYNC
    typedef int v4i __attribute__((vector_size(16)));
    typedef __attribute__((address_space(1))) v4i* gptr_t;   // global (device) AS
    typedef __attribute__((address_space(3))) v4i* lptr_t;   // LDS AS
    __builtin_amdgcn_global_load_async_to_lds_b128((gptr_t)g, (lptr_t)l, 0, 0);
#else
    *(f4*)l = *(const f4*)g;
#endif
}

__device__ __forceinline__ void wait_async_all()
{
#if USE_ASYNC
    __builtin_amdgcn_s_wait_asynccnt(0);
#endif
}

// ---------------- epilogue modes ----------------
enum { M_QK = 0, M_QKV = 1, M_PROJ = 2, M_FC1 = 3, M_FC2 = 4, M_AV = 5 };

__device__ inline float gelu_tanh(float x) {
    float x3 = x * x * x;
    return 0.5f * x * (1.0f + tanhf(0.7978845608028654f * (x + 0.044715f * x3)));
}

// =====================================================================
// Generic bf16 WMMA GEMM: C[M,Ncols] = A[M,K] @ W[Ncols,K]^T  (+ epilogue)
//   BMxBN block tile, BK=32, WAVES_M x WAVES_N waves of 32 lanes.
//   Double-buffered LDS; tile t+1 streams in via async-to-LDS while
//   tile t is consumed by v_wmma_f32_16x16x32_bf16.
// =====================================================================
template<int BM, int BN, int WAVES_M, int WAVES_N, int MODE>
__global__ __launch_bounds__(WAVES_M * WAVES_N * 32)
void gemm_wmma(const __bf16* __restrict__ A, const __bf16* __restrict__ W,
               const float* __restrict__ bias, const float* __restrict__ res,
               float* __restrict__ outF, __bf16* __restrict__ outB,
               __bf16* __restrict__ outB2, __bf16* __restrict__ outB3,
               int M, int Ncols, int K, long batchA, long batchW)
{
    constexpr int BK = 32;
    constexpr int NTHREADS = WAVES_M * WAVES_N * 32;
    constexpr int WM = BM / WAVES_M;          // rows per wave
    constexpr int WN = BN / WAVES_N;          // cols per wave
    constexpr int AF = WM / 16;               // A fragments per wave
    constexpr int BF = WN / 16;               // B fragments per wave
    constexpr int AIT = (BM * BK / 8) / NTHREADS;  // 16B chunks per thread (A)
    constexpr int WIT = (BN * BK / 8) / NTHREADS;  // 16B chunks per thread (W)
    static_assert(AIT >= 1 && WIT >= 1, "tile chunking");
    constexpr int ATILE = BM * BK;
    constexpr int WTILE = BN * BK;

    __shared__ __bf16 sA[2 * ATILE];
    __shared__ __bf16 sW[2 * WTILE];

    const int tid  = threadIdx.x;
    const int wave = tid >> 5;
    const int lane = tid & 31;
    const int wm   = wave / WAVES_N;
    const int wn   = wave % WAVES_N;
    const int lr   = lane & 15;               // 0..15
    const int lh   = lane >> 4;               // 0 / 1

    const int z  = blockIdx.z;
    const __bf16* Ab = A + (size_t)z * (size_t)batchA;
    const __bf16* Wb = W + (size_t)z * (size_t)batchW;

    const int m0 = blockIdx.y * BM;
    const int n0 = blockIdx.x * BN;

    // ---- precompute per-thread staging addresses (uniform trip counts) ----
    const __bf16* aSrc[AIT]; int aOff[AIT];
#pragma unroll
    for (int it = 0; it < AIT; it++) {
        int c  = tid + it * NTHREADS;
        int rr = c / (BK / 8);
        int rc = (c % (BK / 8)) * 8;
        aSrc[it] = Ab + (size_t)(m0 + rr) * K + rc;
        aOff[it] = rr * BK + rc;
    }
    const __bf16* wSrc[WIT]; int wOff[WIT];
#pragma unroll
    for (int it = 0; it < WIT; it++) {
        int c  = tid + it * NTHREADS;
        int rr = c / (BK / 8);
        int rc = (c % (BK / 8)) * 8;
        wSrc[it] = Wb + (size_t)(n0 + rr) * K + rc;
        wOff[it] = rr * BK + rc;
    }

    auto stage = [&](int t, int buf) {
        const size_t koff = (size_t)t * BK;
#pragma unroll
        for (int it = 0; it < AIT; it++)
            async_copy16(aSrc[it] + koff, &sA[buf * ATILE + aOff[it]]);
#pragma unroll
        for (int it = 0; it < WIT; it++)
            async_copy16(wSrc[it] + koff, &sW[buf * WTILE + wOff[it]]);
        // speculative prefetch one K-step further (dropped silently if OOB)
        __builtin_prefetch(aSrc[0] + koff + BK, 0, 1);
        __builtin_prefetch(wSrc[0] + koff + BK, 0, 1);
    };

    v8f acc[AF][BF];
#pragma unroll
    for (int i = 0; i < AF; i++)
#pragma unroll
        for (int j = 0; j < BF; j++)
            acc[i][j] = (v8f){0.f,0.f,0.f,0.f,0.f,0.f,0.f,0.f};

    const int T = K / BK;
    stage(0, 0);

#pragma unroll 1
    for (int t = 0; t < T; ++t) {
        wait_async_all();       // tile t fully landed in LDS (ASYNCcnt in-order)
        __syncthreads();        // all waves see it; all waves done reading buf (t+1)&1
        if (t + 1 < T) stage(t + 1, (t + 1) & 1);   // overlaps with compute below

        const __bf16* bufA = &sA[(t & 1) * ATILE];
        const __bf16* bufW = &sW[(t & 1) * WTILE];

        // ---- build fragments (ISA 16-bit A 16x32 / B 32x16 layouts) ----
        v16bf afr[AF], bfr[BF];
#pragma unroll
        for (int i = 0; i < AF; i++) {
            const __bf16* p = &bufA[(wm * WM + i * 16 + lr) * BK];
            BF16x16 u;
            u.f[0] = *(const f4*)&p[lh * 8];        // K = lh*8 .. +7
            u.f[1] = *(const f4*)&p[16 + lh * 8];   // K = 16+lh*8 .. +7
            afr[i] = u.v;
        }
#pragma unroll
        for (int j = 0; j < BF; j++) {
            const __bf16* p = &bufW[(wn * WN + j * 16 + lr) * BK];
            BF16x16 u;
            u.f[0] = *(const f4*)&p[lh * 8];        // K = lh*8 .. +7
            u.f[1] = *(const f4*)&p[16 + lh * 8];   // K = 16+lh*8 .. +7
            bfr[j] = u.v;
        }
#pragma unroll
        for (int i = 0; i < AF; i++)
#pragma unroll
            for (int j = 0; j < BF; j++)
                acc[i][j] = __builtin_amdgcn_wmma_f32_16x16x32_bf16(
                    false, afr[i], false, bfr[j], (short)0, acc[i][j], false, false);
    }

    // ---- epilogue: C layout — lane n = lr, row m = lh*8 + e ----
#pragma unroll
    for (int i = 0; i < AF; i++)
#pragma unroll
        for (int j = 0; j < BF; j++) {
            const int gcol = n0 + wn * WN + j * 16 + lr;
#pragma unroll
            for (int e = 0; e < 8; e++) {
                const int grow = m0 + wm * WM + i * 16 + lh * 8 + e;
                float v = acc[i][j][e];
                if constexpr (MODE == M_QK) {
                    outF[(size_t)z * N_ * N_ + (size_t)grow * Ncols + gcol] = v;
                } else if constexpr (MODE == M_QKV) {
                    float val = v + bias[gcol];
                    int which = gcol / DIM_;
                    int cc    = gcol % DIM_;
                    int hh    = cc / HD_;
                    int d     = cc % HD_;
                    int b     = grow / N_;
                    int n     = grow % N_;
                    __bf16 bv = (__bf16)val;
                    size_t qidx = (((size_t)b * H_ + hh) * N_ + n) * HD_ + d;
                    if (which == 0)      outB [qidx] = bv;                               // q [B,H,N,HD]
                    else if (which == 1) outB2[qidx] = bv;                               // k [B,H,N,HD]
                    else                 outB3[(((size_t)b * H_ + hh) * HD_ + d) * N_ + n] = bv; // v^T [B,H,HD,N]
                } else if constexpr (MODE == M_PROJ) {
                    size_t o = (size_t)grow * Ncols + gcol;
                    outF[o] = res[o] + v + bias[gcol];
                } else if constexpr (MODE == M_FC1) {
                    outB[(size_t)grow * Ncols + gcol] = (__bf16)gelu_tanh(v + bias[gcol]);
                } else if constexpr (MODE == M_FC2) {
                    size_t o = (size_t)grow * Ncols + gcol;
                    outF[o] = res[o] + v + bias[gcol];
                } else if constexpr (MODE == M_AV) {
                    int b = z / H_, h = z % H_;
                    outB[((size_t)(b * N_ + grow)) * DIM_ + h * HD_ + gcol] = (__bf16)v;
                }
            }
        }
}

// =====================================================================
// LayerNorm: fp32 in -> bf16 out (one block of 256 per row)
// =====================================================================
__global__ __launch_bounds__(256)
void ln_kernel(const float* __restrict__ x, const float* __restrict__ g,
               const float* __restrict__ b, __bf16* __restrict__ out, int dim)
{
    __shared__ float s1[256], s2[256];
    const int row = blockIdx.x;
    const float* xr = x + (size_t)row * dim;
    float a = 0.f, a2 = 0.f;
    for (int i = threadIdx.x; i < dim; i += 256) { float v = xr[i]; a += v; a2 += v * v; }
    s1[threadIdx.x] = a; s2[threadIdx.x] = a2;
    __syncthreads();
    for (int off = 128; off > 0; off >>= 1) {
        if (threadIdx.x < off) {
            s1[threadIdx.x] += s1[threadIdx.x + off];
            s2[threadIdx.x] += s2[threadIdx.x + off];
        }
        __syncthreads();
    }
    const float mean = s1[0] / dim;
    const float var  = s2[0] / dim - mean * mean;
    const float rstd = rsqrtf(var + 1e-5f);
    for (int i = threadIdx.x; i < dim; i += 256) {
        float v = xr[i];
        out[(size_t)row * dim + i] = (__bf16)((v - mean) * rstd * g[i] + b[i]);
    }
}

// =====================================================================
// Row softmax of qk*scale over N_ columns (one block per row)
// =====================================================================
__global__ __launch_bounds__(256)
void softmax_kernel(const float* __restrict__ qk, float* __restrict__ out, float scale)
{
    __shared__ float sred[256];
    const size_t row = blockIdx.x;
    const float* in = qk + row * (size_t)N_;
    float*       o  = out + row * (size_t)N_;
    float vals[N_ / 256];
    float lm = -3.4e38f;
#pragma unroll
    for (int t = 0; t < N_ / 256; t++) {
        vals[t] = in[threadIdx.x + t * 256] * scale;
        lm = fmaxf(lm, vals[t]);
    }
    sred[threadIdx.x] = lm; __syncthreads();
    for (int off = 128; off > 0; off >>= 1) {
        if (threadIdx.x < off) sred[threadIdx.x] = fmaxf(sred[threadIdx.x], sred[threadIdx.x + off]);
        __syncthreads();
    }
    const float m = sred[0];
    __syncthreads();
    float ls = 0.f;
#pragma unroll
    for (int t = 0; t < N_ / 256; t++) { vals[t] = __expf(vals[t] - m); ls += vals[t]; }
    sred[threadIdx.x] = ls; __syncthreads();
    for (int off = 128; off > 0; off >>= 1) {
        if (threadIdx.x < off) sred[threadIdx.x] += sred[threadIdx.x + off];
        __syncthreads();
    }
    const float inv = 1.f / sred[0];
#pragma unroll
    for (int t = 0; t < N_ / 256; t++) o[threadIdx.x + t * 256] = vals[t] * inv;
}

// =====================================================================
// uncertainty = sigmoid(conv_w @ qk_heads + conv_b); attn = attn_mean + u*r
// One thread per (b,n,m); loops over all 12 heads.
// =====================================================================
__global__ __launch_bounds__(256)
void unc_attn_kernel(const float* __restrict__ qk, const float* __restrict__ cw,
                     const float* __restrict__ cb, const float* __restrict__ attn_mean,
                     const float* __restrict__ r, float* __restrict__ unc,
                     __bf16* __restrict__ attn)
{
    __shared__ float scw[H_ * H_];
    __shared__ float scb[H_];
    if (threadIdx.x < H_ * H_) scw[threadIdx.x] = cw[threadIdx.x];
    if (threadIdx.x < H_)      scb[threadIdx.x] = cb[threadIdx.x];
    __syncthreads();

    const size_t idx = (size_t)blockIdx.x * blockDim.x + threadIdx.x;
    const int    m   = (int)(idx & (N_ - 1));
    const size_t t   = idx >> 10;
    const int    n   = (int)(t & (N_ - 1));
    const int    b   = (int)(t >> 10);
    const size_t hs  = (size_t)N_ * N_;
    const size_t base = (((size_t)b * H_) * N_ + n) * (size_t)N_ + m;

    float qh[H_];
#pragma unroll
    for (int h = 0; h < H_; h++) qh[h] = qk[base + h * hs];

#pragma unroll
    for (int g = 0; g < H_; g++) {
        float u = scb[g];
#pragma unroll
        for (int h = 0; h < H_; h++) u += scw[g * H_ + h] * qh[h];
        u = 1.f / (1.f + __expf(-u));
        const size_t o = base + (size_t)g * hs;
        unc[o]  = u;
        attn[o] = (__bf16)(attn_mean[o] + u * r[o]);
    }
}

// =====================================================================
// fp32 -> bf16 weight conversion
// =====================================================================
__global__ __launch_bounds__(256)
void to_bf16_kernel(const float* __restrict__ in, __bf16* __restrict__ out, size_t n)
{
    size_t i = (size_t)blockIdx.x * blockDim.x + threadIdx.x;
    const size_t stride = (size_t)gridDim.x * blockDim.x;
    for (; i < n; i += stride) out[i] = (__bf16)in[i];
}

// =====================================================================
// launch
// =====================================================================
extern "C" void kernel_launch(void* const* d_in, const int* in_sizes, int n_in,
                              void* d_out, int out_size, void* d_ws, size_t ws_size,
                              hipStream_t stream)
{
    const float* x      = (const float*)d_in[0];
    const float* r      = (const float*)d_in[1];
    const float* ln1_g  = (const float*)d_in[2];
    const float* ln1_b  = (const float*)d_in[3];
    const float* qkv_w  = (const float*)d_in[4];
    const float* qkv_b  = (const float*)d_in[5];
    const float* conv_w = (const float*)d_in[6];
    const float* conv_b = (const float*)d_in[7];
    const float* proj_w = (const float*)d_in[8];
    const float* proj_b = (const float*)d_in[9];
    const float* ln2_g  = (const float*)d_in[10];
    const float* ln2_b  = (const float*)d_in[11];
    const float* fc1_w  = (const float*)d_in[12];
    const float* fc1_b  = (const float*)d_in[13];
    const float* fc2_w  = (const float*)d_in[14];
    const float* fc2_b  = (const float*)d_in[15];

    float* out_x  = (float*)d_out;
    float* out_am = out_x  + (size_t)B_ * N_ * DIM_;
    float* out_un = out_am + (size_t)B_ * H_ * N_ * N_;

    const size_t ROWS = (size_t)B_ * N_;           // 2048
    const size_t MAPS = (size_t)B_ * H_ * N_ * N_; // 25,165,824

    char*  ws  = (char*)d_ws;
    size_t off = 0;
    auto alloc = [&](size_t bytes) -> char* {
        char* p = ws + off;
        off += (bytes + 255) & ~(size_t)255;
        return p;
    };
    __bf16* h1     = (__bf16*)alloc(ROWS * DIM_ * 2);
    __bf16* qb     = (__bf16*)alloc(ROWS * DIM_ * 2);
    __bf16* kb     = (__bf16*)alloc(ROWS * DIM_ * 2);
    __bf16* vtb    = (__bf16*)alloc(ROWS * DIM_ * 2);
    float*  qkbuf  = (float*) alloc(MAPS * 4);
    __bf16* attnb  = (__bf16*)alloc(MAPS * 2);
    __bf16* avb    = (__bf16*)alloc(ROWS * DIM_ * 2);
    float*  x1     = (float*) alloc(ROWS * DIM_ * 4);
    __bf16* h2     = (__bf16*)alloc(ROWS * DIM_ * 2);
    __bf16* f1o    = (__bf16*)alloc(ROWS * MLP_ * 2);
    __bf16* qkvwb  = (__bf16*)alloc((size_t)3 * DIM_ * DIM_ * 2);
    __bf16* projwb = (__bf16*)alloc((size_t)DIM_ * DIM_ * 2);
    __bf16* fc1wb  = (__bf16*)alloc((size_t)MLP_ * DIM_ * 2);
    __bf16* fc2wb  = (__bf16*)alloc((size_t)DIM_ * MLP_ * 2);
    (void)ws_size; (void)in_sizes; (void)n_in; (void)out_size;

    // weight conversions
    to_bf16_kernel<<<2048, 256, 0, stream>>>(qkv_w, qkvwb, (size_t)3 * DIM_ * DIM_);
    to_bf16_kernel<<<2048, 256, 0, stream>>>(proj_w, projwb, (size_t)DIM_ * DIM_);
    to_bf16_kernel<<<2048, 256, 0, stream>>>(fc1_w, fc1wb, (size_t)MLP_ * DIM_);
    to_bf16_kernel<<<2048, 256, 0, stream>>>(fc2_w, fc2wb, (size_t)DIM_ * MLP_);

    // LN1
    ln_kernel<<<(int)ROWS, 256, 0, stream>>>(x, ln1_g, ln1_b, h1, DIM_);

    // QKV: [2048,768] @ [2304,768]^T -> split q/k/v^T
    gemm_wmma<128, 128, 2, 4, M_QKV><<<dim3(3 * DIM_ / 128, ROWS / 128, 1), 256, 0, stream>>>(
        h1, qkvwb, qkv_b, nullptr, nullptr, qb, kb, vtb, (int)ROWS, 3 * DIM_, DIM_, 0, 0);

    // QK^T per (b,h): [1024,64] @ [1024,64]^T -> qk fp32
    gemm_wmma<128, 128, 2, 4, M_QK><<<dim3(N_ / 128, N_ / 128, B_ * H_), 256, 0, stream>>>(
        qb, kb, nullptr, nullptr, qkbuf, nullptr, nullptr, nullptr,
        N_, N_, HD_, (long)N_ * HD_, (long)N_ * HD_);

    // softmax -> attn_mean (output region 1)
    softmax_kernel<<<B_ * H_ * N_, 256, 0, stream>>>(qkbuf, out_am, 0.125f);

    // uncertainty (output region 2) + attn (bf16)
    unc_attn_kernel<<<(B_ * N_ * N_) / 256, 256, 0, stream>>>(
        qkbuf, conv_w, conv_b, out_am, r, out_un, attnb);

    // attn @ V per (b,h): [1024,1024] @ [64,1024]^T -> av [B,N,DIM] bf16
    gemm_wmma<128, 64, 4, 2, M_AV><<<dim3(1, N_ / 128, B_ * H_), 256, 0, stream>>>(
        attnb, vtb, nullptr, nullptr, nullptr, avb, nullptr, nullptr,
        N_, HD_, N_, (long)N_ * (long)N_, (long)HD_ * N_);

    // proj + residual -> x1 fp32
    gemm_wmma<128, 128, 2, 4, M_PROJ><<<dim3(DIM_ / 128, ROWS / 128, 1), 256, 0, stream>>>(
        avb, projwb, proj_b, x, x1, nullptr, nullptr, nullptr, (int)ROWS, DIM_, DIM_, 0, 0);

    // LN2
    ln_kernel<<<(int)ROWS, 256, 0, stream>>>(x1, ln2_g, ln2_b, h2, DIM_);

    // FC1 + GELU
    gemm_wmma<128, 128, 2, 4, M_FC1><<<dim3(MLP_ / 128, ROWS / 128, 1), 256, 0, stream>>>(
        h2, fc1wb, fc1_b, nullptr, nullptr, f1o, nullptr, nullptr, (int)ROWS, MLP_, DIM_, 0, 0);

    // FC2 + residual -> final x (output region 0)
    gemm_wmma<128, 128, 2, 4, M_FC2><<<dim3(DIM_ / 128, ROWS / 128, 1), 256, 0, stream>>>(
        f1o, fc2wb, fc2_b, x1, out_x, nullptr, nullptr, nullptr, (int)ROWS, DIM_, MLP_, 0, 0);
}
